// MultiHeadAttention_55920474194163
// MI455X (gfx1250) — matrix operand order repaired and checked by reading, not compiled, not run on hardware
//
#include <hip/hip_runtime.h>
#include <hip/hip_bf16.h>

typedef __attribute__((ext_vector_type(4)))  float    v4f;
typedef __attribute__((ext_vector_type(8)))  float    v8f;
typedef __attribute__((ext_vector_type(4)))  _Float16 v4h;
typedef __attribute__((ext_vector_type(8)))  _Float16 v8h;
typedef __attribute__((ext_vector_type(16))) _Float16 v16h;

#define D_MODEL 1024
#define S_LEN   2048
#define BATCH   2
#define NHEADS  16
#define HDIM    64

// ---------------------------------------------------------------------------
// gfx1250 async global->LDS staging (ASYNCcnt-tracked), with sync fallback
// ---------------------------------------------------------------------------
#if __has_builtin(__builtin_amdgcn_global_load_async_to_lds_b128)
#define HAVE_ASYNC_LDS 1
#else
#define HAVE_ASYNC_LDS 0
#endif

typedef int v4i __attribute__((vector_size(16)));
typedef __attribute__((address_space(1))) v4i as1_v4i;
typedef __attribute__((address_space(3))) v4i as3_v4i;

__device__ __forceinline__ void async_copy16(const void* g, void* l) {
#if HAVE_ASYNC_LDS
  // AS1 pointer value == virtual address; AS3 pointer value == low 32 bits of
  // the generic LDS address.
  __builtin_amdgcn_global_load_async_to_lds_b128(
      (as1_v4i*)(unsigned long long)g,
      (as3_v4i*)(unsigned)(unsigned long long)l, 0, 0);
#else
  *(v8h*)l = *(const v8h*)g;
#endif
}

__device__ __forceinline__ void async_wait_all() {
#if HAVE_ASYNC_LDS
#if __has_builtin(__builtin_amdgcn_s_wait_asynccnt)
  __builtin_amdgcn_s_wait_asynccnt(0);
#else
  asm volatile("s_wait_asynccnt 0x0" ::: "memory");
#endif
#endif
}

__device__ __forceinline__ void lds_wave_fence() {
  // wave-private LDS RAW: DS ops are in-order per wave; just stop the
  // compiler from reordering and drain DScnt.
  asm volatile("s_wait_dscnt 0x0" ::: "memory");
}

// ---------------------------------------------------------------------------
// helpers: build a v16h WMMA fragment from two 16-byte chunks
// ---------------------------------------------------------------------------
__device__ __forceinline__ v16h cat16(v8h lo, v8h hi) {
  return __builtin_shufflevector(lo, hi, 0, 1, 2, 3, 4, 5, 6, 7,
                                         8, 9, 10, 11, 12, 13, 14, 15);
}
// A-matrix fragment (16x32 f16): comps 0..7 at p, comps 8..15 at p+16
__device__ __forceinline__ v16h ldFragA(const _Float16* p) {
  return cat16(*(const v8h*)p, *(const v8h*)(p + 16));
}
// B-matrix fragment (32x16 f16): comps 0..7 at p, comps 8..15 at p+16
__device__ __forceinline__ v16h ldFragB(const _Float16* p) {
  return cat16(*(const v8h*)p, *(const v8h*)(p + 16));
}

// ---------------------------------------------------------------------------
// f32 -> f16 pack
// ---------------------------------------------------------------------------
__global__ void cvt_f32_f16(const float* __restrict__ in,
                            _Float16* __restrict__ out, int n4) {
  int i = blockIdx.x * blockDim.x + threadIdx.x;
  if (i < n4) {
    v4f v = ((const v4f*)in)[i];
    v4h h;
#pragma unroll
    for (int k = 0; k < 4; ++k) h[k] = (_Float16)v[k];
    ((v4h*)out)[i] = h;
  }
}

// ---------------------------------------------------------------------------
// GEMM: Out[M,N] = A[M,K] * W[N,K]^T + bias[N]
// block = 256 threads (8 waves), tile 128(M) x 64(N), K-step 32.
// Double-buffered async global->LDS staging overlaps DMA with WMMA.
// ---------------------------------------------------------------------------
template <bool OUT_HALF>
__global__ __launch_bounds__(256) void gemm_xwT(
    const _Float16* __restrict__ A, const _Float16* __restrict__ W,
    const float* __restrict__ bias, void* __restrict__ Out,
    int M, int N, int K) {
  constexpr int BM = 128, BN = 64, BKK = 32;
  __shared__ __align__(16) _Float16 As[2][BM][40];  // padded: conflict-free
  __shared__ __align__(16) _Float16 Bs[2][BN][40];

  const int tid = threadIdx.x;
  const int wave = tid >> 5, lane = tid & 31;
  const int lane16 = lane & 15, halfw = lane >> 4;
  const int waveM = (wave >> 1) * 32;  // 0,32,64,96
  const int waveN = (wave & 1) * 32;   // 0,32
  const int mBase = blockIdx.y * BM;
  const int nBase = blockIdx.x * BN;

  auto stage = [&](int buf, int k0) {
#pragma unroll
    for (int it = 0; it < 2; ++it) {  // A tile 128x32: 512 x 16B chunks
      int c = tid + it * 256;
      int row = c >> 2, ch = c & 3;
      async_copy16(A + (size_t)(mBase + row) * K + k0 + ch * 8,
                   &As[buf][row][ch * 8]);
    }
    {  // B tile 64x32: 256 chunks
      int row = tid >> 2, ch = tid & 3;
      async_copy16(W + (size_t)(nBase + row) * K + k0 + ch * 8,
                   &Bs[buf][row][ch * 8]);
    }
  };

  v8f acc[2][2] = {};
  const int NT = K / BKK;
  stage(0, 0);
  for (int i = 0; i < NT; ++i) {
    const int cur = i & 1;
    async_wait_all();    // my DMA for buffer `cur` has landed
    __syncthreads();     // everyone's has; prior reads of `cur^1` finished
    if (i + 1 < NT) stage(cur ^ 1, (i + 1) * BKK);
    if (i + 2 < NT) {    // L2 prefetch two tiles ahead
      __builtin_prefetch(A + (size_t)(mBase + (tid >> 1)) * K + (i + 2) * BKK,
                         0, 1);
      __builtin_prefetch(W + (size_t)(nBase + (tid & 63)) * K + (i + 2) * BKK,
                         0, 1);
    }

    v16h aF[2], bF[2];
#pragma unroll
    for (int mi = 0; mi < 2; ++mi)
      aF[mi] = ldFragA(&As[cur][waveM + mi * 16 + lane16][halfw * 8]);
#pragma unroll
    for (int ni = 0; ni < 2; ++ni)
      bF[ni] = ldFragB(&Bs[cur][waveN + ni * 16 + lane16][halfw * 8]);
#pragma unroll
    for (int mi = 0; mi < 2; ++mi)
#pragma unroll
      for (int ni = 0; ni < 2; ++ni)
        acc[mi][ni] = __builtin_amdgcn_wmma_f32_16x16x32_f16(
            false, aF[mi], false, bF[ni], (short)0, acc[mi][ni], false, false);
  }

  // epilogue: C/D layout -> row M = j + 8*halfw, col N = lane16
#pragma unroll
  for (int mi = 0; mi < 2; ++mi)
#pragma unroll
    for (int ni = 0; ni < 2; ++ni)
#pragma unroll
      for (int j = 0; j < 8; ++j) {
        int m = mBase + waveM + mi * 16 + j + 8 * halfw;
        int n = nBase + waveN + ni * 16 + lane16;
        float v = acc[mi][ni][j] + bias[n];
        if constexpr (OUT_HALF)
          ((_Float16*)Out)[(size_t)m * N + n] = (_Float16)v;
        else
          ((float*)Out)[(size_t)m * N + n] = v;
      }
}

// ---------------------------------------------------------------------------
// Flash attention (causal), exp2 domain. block = 128 threads (4 waves),
// one (batch,head, 128-query block) per workgroup; each wave owns 32 rows.
// K tiles async-DMA double-buffered; V transposed through VGPRs (also
// double-buffered); P round-trips through wave-private LDS (no barrier).
// ---------------------------------------------------------------------------
__global__ __launch_bounds__(128) void flash_attn(
    const _Float16* __restrict__ qp, const _Float16* __restrict__ kp,
    const _Float16* __restrict__ vp, _Float16* __restrict__ op) {
  constexpr int BM = 128, BKY = 32;
  __shared__ __align__(16) _Float16 Kt[2][BKY][72];   // [key][dim], padded
  __shared__ __align__(16) _Float16 Vt[2][HDIM][40];  // [dim][key], padded
  __shared__ __align__(16) _Float16 Pt[4][32][40];    // per-wave P

  const int qblk = blockIdx.x;
  const int bh = blockIdx.y;
  const int b = bh / NHEADS, h = bh % NHEADS;
  const int tid = threadIdx.x;
  const int wave = tid >> 5, lane = tid & 31;
  const int lane16 = lane & 15, halfw = lane >> 4;

  const size_t base = ((size_t)b * S_LEN) * D_MODEL + (size_t)h * HDIM;
  const int qRow0 = qblk * BM + wave * 32;

  auto stageKV = [&](int buf, int kRow0) {
#pragma unroll
    for (int it = 0; it < 2; ++it) {  // K tile [32 x 64]
      int c = tid + it * 128;
      int row = c >> 3, ch = c & 7;
      async_copy16(kp + base + (size_t)(kRow0 + row) * D_MODEL + ch * 8,
                   &Kt[buf][row][ch * 8]);
    }
#pragma unroll
    for (int it = 0; it < 2; ++it) {  // V tile transposed -> [dim][key]
      int c = tid + it * 128;
      int row = c >> 3, ch = c & 7;
      v8h vv =
          *(const v8h*)(vp + base + (size_t)(kRow0 + row) * D_MODEL + ch * 8);
#pragma unroll
      for (int i2 = 0; i2 < 8; ++i2) Vt[buf][ch * 8 + i2][row] = vv[i2];
    }
  };

  // Q fragments, pre-scaled by (1/sqrt(Dh)) * log2(e) for exp2-domain softmax
  v16h qa[2][2];
#pragma unroll
  for (int mi = 0; mi < 2; ++mi)
#pragma unroll
    for (int kk = 0; kk < 2; ++kk) {
      const _Float16* p = qp + base +
          (size_t)(qRow0 + mi * 16 + lane16) * D_MODEL + kk * 32 + halfw * 8;
      qa[mi][kk] = ldFragA(p) * (_Float16)0.18033688f;  // 0.125 * log2(e)
    }

  v8f o[2][4] = {};
  float mrow[2][8], lrow[2][8];
#pragma unroll
  for (int mi = 0; mi < 2; ++mi)
#pragma unroll
    for (int j = 0; j < 8; ++j) { mrow[mi][j] = -1e30f; lrow[mi][j] = 0.f; }

  const int nkb = (qblk + 1) * (BM / BKY);
  stageKV(0, 0);
  for (int kb = 0; kb < nkb; ++kb) {
    const int cur = kb & 1;
    const int kRow0 = kb * BKY;
    async_wait_all();
    __syncthreads();
    if (kb + 1 < nkb) stageKV(cur ^ 1, kRow0 + BKY);

    // S = (Q * scale) @ K^T : contraction over 64 dims = 2 WMMA k-steps
    v8f s[2][2] = {};
#pragma unroll
    for (int kk = 0; kk < 2; ++kk)
#pragma unroll
      for (int ni = 0; ni < 2; ++ni) {
        v16h bk = ldFragB(&Kt[cur][ni * 16 + lane16][kk * 32 + halfw * 8]);
#pragma unroll
        for (int mi = 0; mi < 2; ++mi)
          s[mi][ni] = __builtin_amdgcn_wmma_f32_16x16x32_f16(
              false, qa[mi][kk], false, bk, (short)0, s[mi][ni], false, false);
      }

    // causal mask
#pragma unroll
    for (int mi = 0; mi < 2; ++mi)
#pragma unroll
      for (int ni = 0; ni < 2; ++ni) {
        int kg = kRow0 + ni * 16 + lane16;
#pragma unroll
        for (int j = 0; j < 8; ++j) {
          int qg = qRow0 + mi * 16 + j + 8 * halfw;
          if (kg > qg) s[mi][ni][j] = -1e30f;
        }
      }

    // online softmax: row reductions across the 16 lanes of each half-wave
    float rmax[2][8], rsum[2][8], alpha[2][8];
#pragma unroll
    for (int mi = 0; mi < 2; ++mi)
#pragma unroll
      for (int j = 0; j < 8; ++j)
        rmax[mi][j] = fmaxf(s[mi][0][j], s[mi][1][j]);
#pragma unroll
    for (int off = 1; off < 16; off <<= 1)
#pragma unroll
      for (int mi = 0; mi < 2; ++mi)
#pragma unroll
        for (int j = 0; j < 8; ++j)
          rmax[mi][j] = fmaxf(rmax[mi][j], __shfl_xor(rmax[mi][j], off, 32));

#pragma unroll
    for (int mi = 0; mi < 2; ++mi)
#pragma unroll
      for (int j = 0; j < 8; ++j) {
        float mn = fmaxf(mrow[mi][j], rmax[mi][j]);
        alpha[mi][j] = exp2f(mrow[mi][j] - mn);
        mrow[mi][j] = mn;
        rsum[mi][j] = 0.f;
      }
#pragma unroll
    for (int mi = 0; mi < 2; ++mi)
#pragma unroll
      for (int ni = 0; ni < 2; ++ni)
#pragma unroll
        for (int j = 0; j < 8; ++j) {
          float pv = exp2f(s[mi][ni][j] - mrow[mi][j]);
          s[mi][ni][j] = pv;
          rsum[mi][j] += pv;
        }
#pragma unroll
    for (int off = 1; off < 16; off <<= 1)
#pragma unroll
      for (int mi = 0; mi < 2; ++mi)
#pragma unroll
        for (int j = 0; j < 8; ++j)
          rsum[mi][j] += __shfl_xor(rsum[mi][j], off, 32);

#pragma unroll
    for (int mi = 0; mi < 2; ++mi)
#pragma unroll
      for (int j = 0; j < 8; ++j)
        lrow[mi][j] = lrow[mi][j] * alpha[mi][j] + rsum[mi][j];
#pragma unroll
    for (int mi = 0; mi < 2; ++mi)
#pragma unroll
      for (int nd = 0; nd < 4; ++nd)
#pragma unroll
        for (int j = 0; j < 8; ++j) o[mi][nd][j] *= alpha[mi][j];

    // P: C-layout -> wave-private LDS -> A-layout for P@V
#pragma unroll
    for (int mi = 0; mi < 2; ++mi)
#pragma unroll
      for (int ni = 0; ni < 2; ++ni)
#pragma unroll
        for (int j = 0; j < 8; ++j)
          Pt[wave][mi * 16 + j + 8 * halfw][ni * 16 + lane16] =
              (_Float16)s[mi][ni][j];
    lds_wave_fence();  // DS in-order per wave; no block barrier needed

    v16h pa[2];
#pragma unroll
    for (int mi = 0; mi < 2; ++mi)
      pa[mi] = ldFragA(&Pt[wave][mi * 16 + lane16][halfw * 8]);
#pragma unroll
    for (int nd = 0; nd < 4; ++nd) {
      v16h bv = ldFragB(&Vt[cur][nd * 16 + lane16][halfw * 8]);
#pragma unroll
      for (int mi = 0; mi < 2; ++mi)
        o[mi][nd] = __builtin_amdgcn_wmma_f32_16x16x32_f16(
            false, pa[mi], false, bv, (short)0, o[mi][nd], false, false);
    }
  }

  // finalize: divide by row sums, store f16 [b, s, h*64 + d]
#pragma unroll
  for (int mi = 0; mi < 2; ++mi)
#pragma unroll
    for (int nd = 0; nd < 4; ++nd)
#pragma unroll
      for (int j = 0; j < 8; ++j) {
        int row = qRow0 + mi * 16 + j + 8 * halfw;
        int col = nd * 16 + lane16;
        float val = o[mi][nd][j] / lrow[mi][j];
        op[base + (size_t)row * D_MODEL + col] = (_Float16)val;
      }
}

// ---------------------------------------------------------------------------
// launcher
// ---------------------------------------------------------------------------
extern "C" void kernel_launch(void* const* d_in, const int* in_sizes, int n_in,
                              void* d_out, int out_size, void* d_ws,
                              size_t ws_size, hipStream_t stream) {
  (void)in_sizes; (void)n_in; (void)out_size; (void)ws_size;
  const float* Q  = (const float*)d_in[0];
  const float* K  = (const float*)d_in[1];
  const float* V  = (const float*)d_in[2];
  // d_in[3] = mask (causality handled analytically)
  const float* Wq = (const float*)d_in[4];
  const float* bq = (const float*)d_in[5];
  const float* Wk = (const float*)d_in[6];
  const float* bk = (const float*)d_in[7];
  const float* Wv = (const float*)d_in[8];
  const float* bv = (const float*)d_in[9];
  const float* Wo = (const float*)d_in[10];
  const float* bo = (const float*)d_in[11];

  const int M = BATCH * S_LEN;                 // 4096
  const size_t NE = (size_t)M * D_MODEL;       // 4,194,304
  const size_t NW = (size_t)D_MODEL * D_MODEL; // 1,048,576

  _Float16* ws  = (_Float16*)d_ws;
  _Float16* Qh  = ws;            // NE
  _Float16* Kh  = Qh + NE;       // NE
  _Float16* Vh  = Kh + NE;       // NE
  _Float16* Wqh = Vh + NE;       // NW
  _Float16* Wkh = Wqh + NW;
  _Float16* Wvh = Wkh + NW;
  _Float16* Woh = Wvh + NW;
  _Float16* qh  = Woh + NW;      // NE
  _Float16* kh  = qh + NE;       // NE
  _Float16* vh  = kh + NE;       // NE
  _Float16* ah  = Qh;            // attention output reuses Qh (total 56 MB)

  const int T = 256;
  cvt_f32_f16<<<(unsigned)(NE / 4 / T), T, 0, stream>>>(Q, Qh, (int)(NE / 4));
  cvt_f32_f16<<<(unsigned)(NE / 4 / T), T, 0, stream>>>(K, Kh, (int)(NE / 4));
  cvt_f32_f16<<<(unsigned)(NE / 4 / T), T, 0, stream>>>(V, Vh, (int)(NE / 4));
  cvt_f32_f16<<<(unsigned)(NW / 4 / T), T, 0, stream>>>(Wq, Wqh, (int)(NW / 4));
  cvt_f32_f16<<<(unsigned)(NW / 4 / T), T, 0, stream>>>(Wk, Wkh, (int)(NW / 4));
  cvt_f32_f16<<<(unsigned)(NW / 4 / T), T, 0, stream>>>(Wv, Wvh, (int)(NW / 4));
  cvt_f32_f16<<<(unsigned)(NW / 4 / T), T, 0, stream>>>(Wo, Woh, (int)(NW / 4));

  dim3 gg(D_MODEL / 64, M / 128);  // (16, 32)
  gemm_xwT<true><<<gg, 256, 0, stream>>>(Qh, Wqh, bq, qh, M, D_MODEL, D_MODEL);
  gemm_xwT<true><<<gg, 256, 0, stream>>>(Kh, Wkh, bk, kh, M, D_MODEL, D_MODEL);
  gemm_xwT<true><<<gg, 256, 0, stream>>>(Vh, Wvh, bv, vh, M, D_MODEL, D_MODEL);

  flash_attn<<<dim3(S_LEN / 128, BATCH * NHEADS), 128, 0, stream>>>(qh, kh, vh,
                                                                    ah);

  gemm_xwT<false><<<gg, 256, 0, stream>>>(ah, Woh, bo, d_out, M, D_MODEL,
                                          D_MODEL);
}
